// MixNet_38920993636533
// MI455X (gfx1250) — matrix operand order repaired and checked by reading, not compiled, not run on hardware
//
#include <hip/hip_runtime.h>
#include <math.h>

// ---------------- types for WMMA / TDM ----------------
typedef __bf16 bf16_t;
typedef __attribute__((ext_vector_type(8)))  __bf16   v8bf;
typedef __attribute__((ext_vector_type(16))) __bf16   v16bf;
typedef __attribute__((ext_vector_type(8)))  float    v8f;
typedef __attribute__((ext_vector_type(4)))  unsigned v4u;
typedef __attribute__((ext_vector_type(8)))  int      v8i;
typedef __attribute__((ext_vector_type(4)))  int      v4i;

union ABFrag { v16bf v; v8bf h[2]; };

#if defined(__HIP_DEVICE_COMPILE__) && __has_builtin(__builtin_amdgcn_tensor_load_to_lds) && __has_builtin(__builtin_amdgcn_s_wait_tensorcnt)
#define USE_TDM 1
#else
#define USE_TDM 0
#endif

#define NVARS 1024
#define KDIM  32
#define NIN   768
#define BATCH 64
#define SWEEPS 10
#define NBLK  64            // 16-row coordinate blocks covering i = 1..1023

// ---------------- LDS layout (bytes) ----------------
#define OFF_VF     0                    // float  [1024][32]   131072
#define OFF_VT     131072               // bf16   [32][1024]    65536
#define OFF_GPART  196608               // float  [8][16][32]   16384
#define OFF_G      212992               // float  [16][32]       2048
#define OFF_CORR   215040               // float  [16][32]       2048
#define OFF_CBLK   217088               // float  [16][16]       1024
#define OFF_V0     218112               // float  [32]            128
#define OFF_ZF     218240               // float  [1024]         4096
#define OFF_FREE   222336               // int    [1024]         4096
#define OFF_ASTAGE 226432               // bf16   [2][16][1024] 65536 (TDM double buffer)
#define SMEM_BYTES 291968

// ---------------- small helpers ----------------
__device__ __forceinline__ unsigned pcg(unsigned x) {
    x = x * 747796405u + 2891336453u;
    unsigned w = ((x >> ((x >> 28u) + 4u)) ^ x) * 277803737u;
    return (w >> 22u) ^ w;
}
__device__ __forceinline__ float u01(unsigned h) {
    return ((float)(h >> 8) + 0.5f) * (1.0f / 16777216.0f);
}
__device__ __forceinline__ float gauss(unsigned s) {
    unsigned h1 = pcg(s);
    unsigned h2 = pcg(s ^ 0x85EBCA6Bu);
    float u1 = u01(h1), u2 = u01(h2);
    return sqrtf(-2.0f * logf(u1)) * cosf(6.28318530718f * u2);
}
__device__ __forceinline__ float wave_sum(float v) {
#pragma unroll
    for (int m = 16; m >= 1; m >>= 1) v += __shfl_xor(v, m, 32);
    return v;
}

// Issue a TDM 2D tile load: 16 rows x 1024 bf16 from Cb[row0..] into LDS.
// rows_avail = tensor_dim1 so that OOB rows (last partial block) return zeros.
__device__ __forceinline__ void tdm_stage_rows(const bf16_t* gsrc, unsigned lds_off,
                                               int rows_avail) {
#if USE_TDM
    unsigned long long ga = (unsigned long long)(size_t)gsrc;
    v4u g0;
    g0[0] = 1u;                                            // count=1, user desc
    g0[1] = lds_off;                                       // lds_addr (bytes)
    g0[2] = (unsigned)(ga & 0xFFFFFFFFull);                // global_addr[31:0]
    g0[3] = (unsigned)((ga >> 32) & 0x1FFFFFFull)          // global_addr[56:32]
            | (2u << 30);                                  // type = 2 ("image")
    v8i g1;
    g1[0] = (int)(1u << 16);                               // data_size=1 (2 bytes)
    g1[1] = (int)(1024u << 16);                            // tensor_dim0[15:0]=1024
    g1[2] = (int)(((unsigned)rows_avail & 0xFFFFu) << 16); // tensor_dim1[15:0]
    g1[3] = (int)(1024u << 16);                            // tile_dim0 = 1024
    g1[4] = 16;                                            // tile_dim1 = 16, tile_dim2=0
    g1[5] = 1024;                                          // tensor_dim0_stride[31:0]
    g1[6] = 0;                                             // stride hi / dim1_stride
    g1[7] = 0;
    v4i z4 = {0, 0, 0, 0};
#if defined(__clang_major__) && __clang_major__ >= 23
    v8i z8 = {0, 0, 0, 0, 0, 0, 0, 0};
    __builtin_amdgcn_tensor_load_to_lds(g0, g1, z4, z4, z8, 0);
#else
    __builtin_amdgcn_tensor_load_to_lds(g0, g1, z4, z4, 0);
#endif
#else
    (void)gsrc; (void)lds_off; (void)rows_avail;
#endif
}

// ---------------- kernel 1: C f32 -> bf16 ----------------
__global__ void cvt_c_kernel(const float* __restrict__ C, bf16_t* __restrict__ Cb, int n) {
    int i = blockIdx.x * 256 + threadIdx.x;
    if (i < n) Cb[i] = (bf16_t)C[i];
}

// ---------------- kernel 2: per-batch mixing solver ----------------
__global__ void __launch_bounds__(256)
mixnet_kernel(const float* __restrict__ C,       // [1024][1024] f32 (diag blocks)
              const float* __restrict__ z,       // [64][768]
              const int*   __restrict__ isin,    // [64][768]
              const bf16_t* __restrict__ Cb,     // [1024][1024] bf16
              float* __restrict__ out)           // [64][768]
{
    extern __shared__ char smem[];
    float*  Vf     = (float*) (smem + OFF_VF);     // master V, f32 [n][k]
    bf16_t* Vt     = (bf16_t*)(smem + OFF_VT);     // transposed mirror [k][n]
    float*  Gpart  = (float*) (smem + OFF_GPART);
    float*  G      = (float*) (smem + OFF_G);
    float*  corr   = (float*) (smem + OFF_CORR);
    float*  Cblk   = (float*) (smem + OFF_CBLK);
    float*  v0s    = (float*) (smem + OFF_V0);
    float*  zf     = (float*) (smem + OFF_ZF);
    int*    freeS  = (int*)   (smem + OFF_FREE);
    bf16_t* Astage = (bf16_t*)(smem + OFF_ASTAGE); // [2][16][1024] bf16

    const int b    = blockIdx.x;
    const int tid  = threadIdx.x;
    const int lane = tid & 31;
    const int wave = tid >> 5;
    const int lo   = lane & 15;
    const int hi   = lane >> 4;

    // ---- phase A: z_full + free mask ----
    for (int i = tid; i < NVARS; i += 256) {
        float zv; int fr;
        if (i == 0)            { zv = 1.0f; fr = 0; }
        else if (i <= NIN)     { zv = z[(size_t)b * NIN + (i - 1)];
                                 fr = (isin[(size_t)b * NIN + (i - 1)] == 0); }
        else                   { zv = 0.0f; fr = 1; }
        zf[i] = zv; freeS[i] = fr;
    }

    // ---- phase B: v0 (wave 0) ----
    if (wave == 0) {
        float n0 = gauss(pcg(0xC0FFEEu ^ (unsigned)b) ^ (lane * 0x9E3779B9u));
        float ss = wave_sum(n0 * n0);
        float v0 = n0 * rsqrtf(fmaxf(ss, 1e-24f));
        v0s[lane] = v0;
    }
    __syncthreads();

    // ---- phase C: init V rows ----
    {
        const float v0k = v0s[lane];
        for (int n = wave; n < NVARS; n += 8) {
            float r = gauss(pcg((unsigned)(b * NVARS + n)) ^ (lane * 0x9E3779B9u));
            float dot = wave_sum(r * v0k);
            float r2 = r - dot * v0k;
            float ss = wave_sum(r2 * r2);
            float R = r2 * rsqrtf(fmaxf(ss, 1e-24f));
            float v;
            if (n == 0) v = v0k;
            else if (freeS[n]) v = R;
            else {
                float c = cosf(3.14159265358979f * zf[n]);
                float s = sinf(3.14159265358979f * zf[n]);
                v = -c * v0k + s * R;
            }
            Vf[n * KDIM + lane] = v;
            Vt[lane * NVARS + n] = (bf16_t)v;
        }
    }
    __syncthreads();

#if USE_TDM
    const unsigned astage_off = (unsigned)(size_t)(smem + OFF_ASTAGE);
    // pre-stage step 0 (rows 1..16) into buffer 0
    if (wave == 0) tdm_stage_rows(Cb + (size_t)1 * NVARS, astage_off, NVARS - 1);
#endif

    // ---- mixing sweeps: blocked Gauss-Seidel, WMMA bulk + f32 corrections ----
    const int TOTAL = SWEEPS * NBLK;
    for (int step = 0; step < TOTAL; ++step) {
        const int blk    = step & (NBLK - 1);
        const int i0     = 1 + blk * 16;
        const int parity = step & 1;

        // stage 16x16 diagonal block of C (f32), prefetch next one, zero corr
        {
            int r = tid >> 4, c = tid & 15;
            int rr = i0 + r; if (rr > NVARS - 1) rr = NVARS - 1;
            int cc = i0 + c; if (cc > NVARS - 1) cc = NVARS - 1;
            Cblk[r * 16 + c] = C[(size_t)rr * NVARS + cc];
            int ni0 = 1 + ((step + 1) & (NBLK - 1)) * 16;
            int nrr = ni0 + r; if (nrr > NVARS - 1) nrr = NVARS - 1;
            __builtin_prefetch(&C[(size_t)nrr * NVARS + ni0], 0, 3);
        }
        for (int e = tid; e < 512; e += 256) corr[e] = 0.0f;

#if USE_TDM
        // async-stage next block's A rows into the other buffer, then make sure
        // this block's staging has landed (TDM completes in order per wave)
        if (wave == 0) {
            if (step + 1 < TOTAL) {
                int ni0 = 1 + ((step + 1) & (NBLK - 1)) * 16;
                tdm_stage_rows(Cb + (size_t)ni0 * NVARS,
                               astage_off + (unsigned)((parity ^ 1) * 32768),
                               NVARS - ni0);
                __builtin_amdgcn_s_wait_tensorcnt(1);
            } else {
                __builtin_amdgcn_s_wait_tensorcnt(0);
            }
        }
#endif
        __syncthreads();

        // bulk G = C_blk (16x1024) x V (1024x32), K split over 8 waves
        {
#if USE_TDM
            const bf16_t* Crow = Astage + (size_t)parity * 16384 + (size_t)lo * NVARS;
#else
            int arow = i0 + lo; if (arow > NVARS - 1) arow = NVARS - 1;
            const bf16_t* Crow = Cb + (size_t)arow * NVARS;
#endif
            v8f acc0 = {}; v8f acc1 = {};
#pragma unroll
            for (int j = 0; j < 4; ++j) {
                const int kk = wave * 128 + j * 32;
                ABFrag a;
                a.h[0] = *(const v8bf*)(Crow + kk + hi * 16);
                a.h[1] = *(const v8bf*)(Crow + kk + hi * 16 + 8);
                v16bf b0 = *(const v16bf*)(Vt + (size_t)lo        * NVARS + kk + hi * 16);
                v16bf b1 = *(const v16bf*)(Vt + (size_t)(16 + lo) * NVARS + kk + hi * 16);
                acc0 = __builtin_amdgcn_wmma_f32_16x16x32_bf16(
                    false, a.v, false, b0, (short)0, acc0, false, false);
                acc1 = __builtin_amdgcn_wmma_f32_16x16x32_bf16(
                    false, a.v, false, b1, (short)0, acc1, false, false);
            }
#if !USE_TDM
            if (step + 1 < TOTAL) {
                int nrow = 1 + ((step + 1) & (NBLK - 1)) * 16 + lo;
                if (nrow > NVARS - 1) nrow = NVARS - 1;
                __builtin_prefetch(Cb + (size_t)nrow * NVARS + wave * 128, 0, 3);
            }
#endif
#pragma unroll
            for (int r = 0; r < 8; ++r) {
                const int M = r + 8 * hi;
                Gpart[wave * 512 + M * 32 + lo]      = acc0[r];
                Gpart[wave * 512 + M * 32 + 16 + lo] = acc1[r];
            }
        }
        __syncthreads();

        // reduce 8 partials
        for (int e = tid; e < 512; e += 256) {
            float s = 0.0f;
#pragma unroll
            for (int w = 0; w < 8; ++w) s += Gpart[w * 512 + e];
            G[e] = s;
        }
        __syncthreads();

        // sequential intra-block Gauss-Seidel (wave 0; lane == K component)
        if (wave == 0) {
#pragma unroll 1
            for (int t = 0; t < 16; ++t) {
                const int i = i0 + t;
                if (i < NVARS) {
                    float g = G[t * 32 + lane] + corr[t * 32 + lane];
                    float ss = wave_sum(g * g);
                    float gn = fmaxf(sqrtf(ss), 1e-12f);
                    float vold = Vf[i * KDIM + lane];
                    float vnew = freeS[i] ? (-g / gn) : vold;
                    float delta = vnew - vold;
                    Vf[i * KDIM + lane] = vnew;
                    Vt[lane * NVARS + i] = (bf16_t)vnew;
#pragma unroll 1
                    for (int t2 = t + 1; t2 < 16; ++t2)
                        corr[t2 * 32 + lane] += Cblk[t2 * 16 + t] * delta;
                }
            }
        }
        __syncthreads();
    }

    // ---- readout: z_out = free ? arccos(clip(-V.v0))/pi : z_full, cols 1..768 ----
    {
        const float v0k = v0s[lane];
        for (int n = 1 + wave; n <= NIN; n += 8) {
            float p = Vf[n * KDIM + lane] * v0k;
            float s = wave_sum(p);
            float ca = -s;
            ca = fminf(fmaxf(ca, -1.0f + 1e-7f), 1.0f - 1e-7f);
            float zo = freeS[n] ? (acosf(ca) * (1.0f / 3.14159265358979f)) : zf[n];
            if (lane == 0) out[(size_t)b * NIN + (n - 1)] = zo;
        }
    }
}

// ---------------- host entry ----------------
extern "C" void kernel_launch(void* const* d_in, const int* in_sizes, int n_in,
                              void* d_out, int out_size, void* d_ws, size_t ws_size,
                              hipStream_t stream) {
    (void)in_sizes; (void)n_in; (void)out_size; (void)ws_size;
    const float* C    = (const float*)d_in[0];
    const float* z    = (const float*)d_in[1];
    const int*   isin = (const int*)  d_in[2];
    bf16_t*      Cb   = (bf16_t*)d_ws;          // 1024*1024*2 = 2 MB scratch

    const int nC = NVARS * NVARS;
    cvt_c_kernel<<<(nC + 255) / 256, 256, 0, stream>>>(C, Cb, nC);
    mixnet_kernel<<<BATCH, 256, SMEM_BYTES, stream>>>(C, z, isin, Cb, (float*)d_out);
}
